// NBInjector_67224828117372
// MI455X (gfx1250) — compile-verified, matrix operand order checked by reading, not run on hardware
//
#include <hip/hip_runtime.h>
#include <hip/hip_bf16.h>
#include <math.h>

// ---------------- problem constants ----------------
constexpr int Bd   = 4;
constexpr int Pd   = 576;
constexpr int Cd   = 768;
constexpr int Vd   = 50000;
constexpr int DNB  = 300;
constexpr int Kt   = 3;          // top-k
constexpr int Hd   = 768;
constexpr int IN_DIM = Kt * DNB + Kt;   // 903
constexpr int INP    = IN_DIM + 1;      // 904 (padded K for f32 WMMA)
constexpr int Md   = Bd * Pd;           // 2304 rows
constexpr int MTILES = Md / 16;         // 144 (phase-2 GEMMs)
constexpr int MB     = 64;              // phase-1 M-rows per block (B-fragment reuse x4)
constexpr int MT64   = Md / MB;         // 36
constexpr int CHUNK  = 2048;            // N cols per block in phase 1
constexpr int NCHUNK = 25;              // 25 * 2048 = 51200
constexpr int NPAD   = CHUNK * NCHUNK;  // 51200 (>= Vd, zero padded)

// ---------------- vector types ----------------
typedef __attribute__((ext_vector_type(2)))  float  v2f;
typedef __attribute__((ext_vector_type(8)))  float  v8f;
typedef __attribute__((ext_vector_type(8)))  __bf16 v8bf;
typedef __attribute__((ext_vector_type(16))) __bf16 v16bf;

// ---------------- helpers ----------------
__device__ __forceinline__ v16bf load_bf16_frag(const __bf16* p) {
    // ISA 16-bit A/B layout: this lane holds K = khalf+0..7 and K = khalf+16..23
    v8bf lo = *(const v8bf*)(p);
    v8bf hi = *(const v8bf*)(p + 16);
    v16bf r;
#pragma unroll
    for (int i = 0; i < 8; ++i) { r[i] = lo[i]; r[i + 8] = hi[i]; }
    return r;
}

__device__ __forceinline__ void insert3(float* v, int* ix, float nv, int ni) {
    if (nv > v[2]) {
        if (nv > v[0])      { v[2]=v[1]; ix[2]=ix[1]; v[1]=v[0]; ix[1]=ix[0]; v[0]=nv; ix[0]=ni; }
        else if (nv > v[1]) { v[2]=v[1]; ix[2]=ix[1]; v[1]=nv;  ix[1]=ni; }
        else                { v[2]=nv;  ix[2]=ni; }
    }
}

// ================= kernel 1: T_clip -> bf16 (padded to NPAD rows) =================
__global__ __launch_bounds__(256)
void tconv_kernel(const float* __restrict__ T, __bf16* __restrict__ Tb) {
    size_t e = (size_t)blockIdx.x * 256 + threadIdx.x;
    if (e >= (size_t)NPAD * Cd) return;
    size_t r = e / Cd;
    Tb[e] = (r < (size_t)Vd) ? (__bf16)T[e] : (__bf16)0.0f;
}

// ================= kernel 2: normalize v_seq rows -> bf16 q =================
__global__ __launch_bounds__(256)
void qnorm_kernel(const float* __restrict__ v_seq, __bf16* __restrict__ qb) {
    __shared__ float red[256];
    int row = blockIdx.x, tid = threadIdx.x;
    const float* vr = v_seq + (size_t)row * Cd;
    float s = 0.f;
    for (int j = tid; j < Cd; j += 256) { float x = vr[j]; s += x * x; }
    red[tid] = s; __syncthreads();
    for (int w = 128; w > 0; w >>= 1) { if (tid < w) red[tid] += red[tid + w]; __syncthreads(); }
    float scale = 1.f / fmaxf(sqrtf(red[0]), 1e-12f);
    for (int j = tid; j < Cd; j += 256)
        qb[(size_t)row * Cd + j] = (__bf16)(vr[j] * scale);
}

// ================= kernel 3: bf16 WMMA similarity GEMM fused with top-3 =================
// Block tile: 64 M-rows x 2048 N-cols.  A tile staged in LDS (96 KB); each wave
// keeps 4 accumulators so one streamed B-fragment feeds 4 WMMAs (4x L2-traffic cut).
__global__ __launch_bounds__(256)
void sim_topk_kernel(const __bf16* __restrict__ qb, const __bf16* __restrict__ Tb,
                     float* __restrict__ pval, int* __restrict__ pidx) {
    __shared__ __bf16 Atile[MB * Cd];            // 98,304 B
    __shared__ float  Stile[MB][132];            // 33,792 B
    const int tid  = threadIdx.x;
    const int wave = tid >> 5, lane = tid & 31;
    const int mbase = blockIdx.x * MB;
    const int chunk = blockIdx.y;
    const int nbase_chunk = chunk * CHUNK;

    // ---- stage A tile (64 x 768 bf16) into LDS, 16B per thread per step ----
    for (int idx = tid; idx < MB * Cd / 8; idx += 256) {
        int r = idx / (Cd / 8);
        int c = (idx - r * (Cd / 8)) * 8;
        *(v8bf*)(&Atile[r * Cd + c]) = *(const v8bf*)(qb + (size_t)(mbase + r) * Cd + c);
    }
    __syncthreads();

    const int lrow  = lane & 15;
    const int khalf = (lane >> 4) << 3;          // 0 or 8 (bf16 fragment K offset)

    const int srow = tid >> 2;                   // scan row 0..63
    const int ssub = tid & 3;                    // scan stripe 0..3 (32 cols each)

    float bv[3] = {-3.0e38f, -3.0e38f, -3.0e38f};
    int   bi[3] = {-1, -1, -1};

    for (int s = 0; s < CHUNK / 128; ++s) {
        const int ntile = nbase_chunk + s * 128 + wave * 16;
        const __bf16* brow = Tb + (size_t)(ntile + lrow) * Cd + khalf;
        if (ntile + 128 + lrow < NPAD)           // prefetch next sub-step's B row
            __builtin_prefetch(brow + (size_t)128 * Cd, 0, 1);

        v8f acc[4] = {{}, {}, {}, {}};
#pragma unroll 2
        for (int k = 0; k < Cd; k += 32) {
            v16bf bf = load_bf16_frag(brow + k);
#pragma unroll
            for (int mt = 0; mt < 4; ++mt) {
                v16bf af = load_bf16_frag(&Atile[(mt * 16 + lrow) * Cd + k + khalf]);
                acc[mt] = __builtin_amdgcn_wmma_f32_16x16x32_bf16(
                    false, af, false, bf, (short)0, acc[mt], false, false);
            }
        }
        // C layout: vgpr r, lane -> row r + 8*(lane>>4), col lane&15
        const int chalf = lane >> 4;
        const int ccol  = wave * 16 + (lane & 15);
#pragma unroll
        for (int mt = 0; mt < 4; ++mt)
#pragma unroll
            for (int r = 0; r < 8; ++r)
                Stile[mt * 16 + r + 8 * chalf][ccol] = acc[mt][r];
        __syncthreads();

        // 4 threads per row scan 32 cols each, keep running top-3
        const int base = ssub * 32;
        for (int c = 0; c < 32; ++c) {
            int gcol = nbase_chunk + s * 128 + base + c;
            float vx = (gcol < Vd) ? Stile[srow][base + c] : -3.0e38f;
            insert3(bv, bi, vx, gcol);
        }
        __syncthreads();
    }

    // merge the 4 stripes of this row (contiguous 4-lane group of the wave)
    for (int m = 2; m >= 1; m >>= 1) {
        float ov[3]; int oi[3];
#pragma unroll
        for (int t = 0; t < 3; ++t) {
            ov[t] = __shfl_xor(bv[t], m, 32);
            oi[t] = __shfl_xor(bi[t], m, 32);
        }
#pragma unroll
        for (int t = 0; t < 3; ++t) insert3(bv, bi, ov[t], oi[t]);
    }
    if (ssub == 0) {
        int grow = mbase + srow;
#pragma unroll
        for (int t = 0; t < 3; ++t) {
            pval[((size_t)grow * NCHUNK + chunk) * 3 + t] = bv[t];
            pidx[((size_t)grow * NCHUNK + chunk) * 3 + t] = bi[t];
        }
    }
}

// ================= kernel 4: merge per-chunk top-3 -> final top-3 =================
__global__ __launch_bounds__(256)
void topk_merge_kernel(const float* __restrict__ pval, const int* __restrict__ pidx,
                       float* __restrict__ fval, int* __restrict__ fidx) {
    int row = blockIdx.x * 256 + threadIdx.x;
    if (row >= Md) return;
    float bv[3] = {-3.0e38f, -3.0e38f, -3.0e38f};
    int   bi[3] = {0, 0, 0};
    for (int c = 0; c < NCHUNK; ++c)
#pragma unroll
        for (int t = 0; t < 3; ++t)
            insert3(bv, bi, pval[((size_t)row * NCHUNK + c) * 3 + t],
                            pidx[((size_t)row * NCHUNK + c) * 3 + t]);
#pragma unroll
    for (int t = 0; t < 3; ++t) {
        fval[row * 3 + t] = bv[t];
        fidx[row * 3 + t] = bi[t];
    }
}

// ================= kernel 5: gather + concat + affine LayerNorm -> xn (padded K=904) =================
__global__ __launch_bounds__(256)
void fuse_ln_kernel(const float* __restrict__ nb_vecs, const float* __restrict__ fval,
                    const int* __restrict__ fidx, const float* __restrict__ ln_w,
                    const float* __restrict__ ln_b, float* __restrict__ xn) {
    __shared__ float xbuf[IN_DIM];
    __shared__ float r1[256], r2[256];
    int row = blockIdx.x, tid = threadIdx.x;
    float s1 = 0.f, s2 = 0.f;
    for (int j = tid; j < IN_DIM; j += 256) {
        float x;
        if (j < Kt * DNB) {
            int kk = j / DNB, d = j - kk * DNB;
            int id = fidx[row * 3 + kk]; if (id < 0) id = 0;
            x = nb_vecs[(size_t)id * DNB + d];
        } else {
            x = fval[row * 3 + (j - Kt * DNB)];
        }
        xbuf[j] = x; s1 += x; s2 += x * x;
    }
    r1[tid] = s1; r2[tid] = s2; __syncthreads();
    for (int w = 128; w > 0; w >>= 1) {
        if (tid < w) { r1[tid] += r1[tid + w]; r2[tid] += r2[tid + w]; }
        __syncthreads();
    }
    float mu   = r1[0] * (1.f / IN_DIM);
    float var  = r2[0] * (1.f / IN_DIM) - mu * mu;
    float rstd = rsqrtf(var + 1e-5f);
    for (int j = tid; j < IN_DIM; j += 256)
        xn[(size_t)row * INP + j] = (xbuf[j] - mu) * rstd * ln_w[j] + ln_b[j];
    if (tid == 0) xn[(size_t)row * INP + IN_DIM] = 0.f;
}

// ================= kernel 6: pad W1 rows 903 -> 904 =================
__global__ __launch_bounds__(256)
void w1pad_kernel(const float* __restrict__ W1, float* __restrict__ W1p) {
    size_t e = (size_t)blockIdx.x * 256 + threadIdx.x;
    if (e >= (size_t)Hd * INP) return;
    size_t r = e / INP, c = e - r * INP;
    W1p[e] = (c < (size_t)IN_DIM) ? W1[r * IN_DIM + c] : 0.f;
}

// ================= kernel 7/8: f32 WMMA GEMM  C = act(A*Bw^T + bias) =================
__global__ __launch_bounds__(256)
void gemm_f32_kernel(const float* __restrict__ A, int lda,
                     const float* __restrict__ Bw, int ldb,
                     const float* __restrict__ bias,
                     float* __restrict__ Cout, int ldc, int Kdim, int gelu) {
    const int tid = threadIdx.x, wave = tid >> 5, lane = tid & 31;
    const int mbase = blockIdx.x * 16;
    const int ntile = blockIdx.y * 128 + wave * 16;
    const int lrow  = lane & 15;
    const int koff  = (lane >> 4) * 2;                // f32 16x16x4 fragment K offset
    const float* arow = A  + (size_t)(mbase + lrow) * lda + koff;
    const float* brow = Bw + (size_t)(ntile + lrow) * ldb + koff;

    v8f acc = {};
#pragma unroll 4
    for (int k = 0; k < Kdim; k += 4) {
        v2f a = *(const v2f*)(arow + k);
        v2f b = *(const v2f*)(brow + k);
        acc = __builtin_amdgcn_wmma_f32_16x16x4_f32(
            false, a, false, b, (short)0, acc, false, false);
    }
    const int chalf = lane >> 4, ccol = lane & 15;
#pragma unroll
    for (int r = 0; r < 8; ++r) {
        int m = mbase + r + 8 * chalf;
        int n = ntile + ccol;
        float x = acc[r] + bias[n];
        if (gelu) x = 0.5f * x * (1.0f + erff(x * 0.70710678118654752f));
        Cout[(size_t)m * ldc + n] = x;
    }
}

// ================= kernel 9: gate GEMM (K=1536) + sigmoid + residual + final LN =================
__global__ __launch_bounds__(256)
void gate_ln_kernel(const float* __restrict__ v_seq, const float* __restrict__ nb_feat,
                    const float* __restrict__ gate_W, const float* __restrict__ gate_b,
                    float* __restrict__ out) {
    __shared__ float tmp[16][772];
    const int tid = threadIdx.x, wave = tid >> 5, lane = tid & 31;
    const int mbase = blockIdx.x * 16;
    const int lrow  = lane & 15;
    const int koff  = (lane >> 4) * 2;
    const int am    = mbase + lrow;
    const float* vrow = v_seq   + (size_t)am * Cd;
    const float* frow = nb_feat + (size_t)am * Cd;

    for (int j = 0; j < 6; ++j) {                    // 8 waves * 6 = 48 n-tiles = 768 cols
        const int ntile = (wave + 8 * j) * 16;
        const float* brow = gate_W + (size_t)(ntile + lrow) * (2 * Cd) + koff;
        v8f acc = {};
#pragma unroll 4
        for (int k = 0; k < 2 * Cd; k += 4) {
            int kk = k + koff;                        // fragment never straddles the concat seam
            v2f a = (kk < Cd) ? *(const v2f*)(vrow + kk)
                              : *(const v2f*)(frow + (kk - Cd));
            v2f b = *(const v2f*)(brow + k);
            acc = __builtin_amdgcn_wmma_f32_16x16x4_f32(
                false, a, false, b, (short)0, acc, false, false);
        }
        const int chalf = lane >> 4, ccol = lane & 15;
#pragma unroll
        for (int r = 0; r < 8; ++r) {
            int m = r + 8 * chalf;                    // local row
            int n = ntile + ccol;
            float u  = acc[r] + gate_b[n];
            float g  = 1.f / (1.f + expf(-u));
            float vv = v_seq[(size_t)(mbase + m) * Cd + n];
            float nf = nb_feat[(size_t)(mbase + m) * Cd + n];
            tmp[m][n] = vv + g * nf;
        }
    }
    __syncthreads();

    // final LayerNorm (no affine), one 16-lane half per row
    const int srow = tid >> 4, ssub = tid & 15;
    float s1 = 0.f, s2 = 0.f;
    for (int c = ssub; c < Cd; c += 16) { float x = tmp[srow][c]; s1 += x; s2 += x * x; }
    for (int m = 8; m >= 1; m >>= 1) { s1 += __shfl_xor(s1, m, 32); s2 += __shfl_xor(s2, m, 32); }
    float mu   = s1 * (1.f / Cd);
    float var  = s2 * (1.f / Cd) - mu * mu;
    float rstd = rsqrtf(var + 1e-5f);
    for (int c = ssub; c < Cd; c += 16)
        out[(size_t)(mbase + srow) * Cd + c] = (tmp[srow][c] - mu) * rstd;
}

// ================= launcher =================
extern "C" void kernel_launch(void* const* d_in, const int* in_sizes, int n_in,
                              void* d_out, int out_size, void* d_ws, size_t ws_size,
                              hipStream_t stream) {
    (void)in_sizes; (void)n_in; (void)out_size; (void)ws_size;
    const float* v_seq  = (const float*)d_in[0];
    const float* T_clip = (const float*)d_in[1];
    const float* nb_vec = (const float*)d_in[2];
    const float* ln_w   = (const float*)d_in[3];
    const float* ln_b   = (const float*)d_in[4];
    const float* W1     = (const float*)d_in[5];
    const float* b1     = (const float*)d_in[6];
    const float* W2     = (const float*)d_in[7];
    const float* b2     = (const float*)d_in[8];
    const float* gate_W = (const float*)d_in[9];
    const float* gate_b = (const float*)d_in[10];
    float* out = (float*)d_out;
    char*  ws  = (char*)d_ws;

    // --- workspace layout (bytes, 256-aligned) ---
    constexpr size_t SZ_TB   = (size_t)NPAD * Cd * 2;          // 78,643,200
    constexpr size_t OFF_TB  = 0;
    constexpr size_t OFF_QB  = OFF_TB + SZ_TB;
    constexpr size_t SZ_QB   = (size_t)Md * Cd * 2;
    constexpr size_t OFF_PV  = OFF_QB + SZ_QB;
    constexpr size_t SZ_P    = (size_t)Md * NCHUNK * 3 * 4;
    constexpr size_t OFF_PI  = OFF_PV + SZ_P;
    constexpr size_t OFF_FV  = OFF_PI + SZ_P;
    constexpr size_t SZ_F    = (size_t)Md * 3 * 4;
    constexpr size_t OFF_FI  = OFF_FV + SZ_F;
    // phase-2 buffers reuse the (dead after top-k) Tb region
    constexpr size_t OFF_XN  = 0;                               // Md*INP*4  = 8,331,264
    constexpr size_t OFF_W1P = OFF_XN  + (size_t)Md * INP * 4;  // Hd*INP*4  = 2,777,088
    constexpr size_t OFF_H   = OFF_W1P + (size_t)Hd * INP * 4;  // Md*Hd*4   = 7,077,888
    constexpr size_t OFF_NBF = OFF_H   + (size_t)Md * Hd * 4;   // Md*Cd*4
    static_assert(OFF_NBF + (size_t)Md * Cd * 4 <= SZ_TB, "reuse region overflow");

    __bf16* Tb   = (__bf16*)(ws + OFF_TB);
    __bf16* qb   = (__bf16*)(ws + OFF_QB);
    float*  pval = (float*) (ws + OFF_PV);
    int*    pidx = (int*)   (ws + OFF_PI);
    float*  fval = (float*) (ws + OFF_FV);
    int*    fidx = (int*)   (ws + OFF_FI);
    float*  xn   = (float*) (ws + OFF_XN);
    float*  W1p  = (float*) (ws + OFF_W1P);
    float*  h    = (float*) (ws + OFF_H);
    float*  nbf  = (float*) (ws + OFF_NBF);

    // 1-2: precision staging (T fits L2 in bf16: 76.8 MB < 192 MB)
    tconv_kernel<<<(unsigned)(((size_t)NPAD * Cd) / 256), 256, 0, stream>>>(T_clip, Tb);
    qnorm_kernel<<<Md, 256, 0, stream>>>(v_seq, qb);

    // 3-4: fused similarity GEMM + top-3 (bf16 WMMA, 64-row blocks), then merge
    sim_topk_kernel<<<dim3(MT64, NCHUNK), 256, 0, stream>>>(qb, Tb, pval, pidx);
    topk_merge_kernel<<<(Md + 255) / 256, 256, 0, stream>>>(pval, pidx, fval, fidx);

    // 5-6: build + normalize features, pad W1
    fuse_ln_kernel<<<Md, 256, 0, stream>>>(nb_vec, fval, fidx, ln_w, ln_b, xn);
    w1pad_kernel<<<(unsigned)(((size_t)Hd * INP + 255) / 256), 256, 0, stream>>>(W1, W1p);

    // 7-8: MLP in f32 WMMA (16x16x4)
    gemm_f32_kernel<<<dim3(MTILES, Hd / 128), 256, 0, stream>>>(xn, INP, W1p, INP, b1, h, Hd, INP, 1);
    gemm_f32_kernel<<<dim3(MTILES, Cd / 128), 256, 0, stream>>>(h, Hd, W2, Hd, b2, nbf, Cd, Hd, 0);

    // 9: gate + residual + final LN
    gate_ln_kernel<<<MTILES, 256, 0, stream>>>(v_seq, nbf, gate_W, gate_b, out);
}
